// ConditionedVSSBlock_33457795235972
// MI455X (gfx1250) — matrix operand order repaired and checked by reading, not compiled, not run on hardware
//
#include <hip/hip_runtime.h>
#include <hip/hip_bf16.h>

// ---------------------------------------------------------------------------
// Dims (compile-time, from reference)
// ---------------------------------------------------------------------------
#define BB   4
#define HH_  64
#define WW_  64
#define LL   4096          // H*W
#define DM   256           // d_model
#define DI   512           // d_inner
#define NS   16            // d_state
#define DTR  16            // dt_rank
#define KK   4             // directions
#define CPROJ 48           // dt_rank + 2*d_state
#define MROWS (BB*LL)      // 16384

typedef __attribute__((ext_vector_type(16))) _Float16 v16h;
typedef __attribute__((ext_vector_type(8)))  float    v8f;
typedef __attribute__((ext_vector_type(4)))  unsigned u32x4;
typedef __attribute__((ext_vector_type(8)))  unsigned u32x8;

union HFrag { v16h v; unsigned u[8]; };

// ---------------------------------------------------------------------------
// WMMA fragment loaders (k = 16*(lane/16) + s in both A and B, wave32)
// ---------------------------------------------------------------------------
__device__ __forceinline__ v16h load_a_frag(const _Float16* sA, int mbase, int lane) {
  HFrag f;
  int m  = lane & 15;
  int hq = (lane >> 4) * 8;                     // u32 offset for K subgroup
  const unsigned* rp = (const unsigned*)(sA + (mbase + m) * 32);
#pragma unroll
  for (int p = 0; p < 8; ++p) f.u[p] = rp[hq + p];
  return f.v;
}
__device__ __forceinline__ v16h load_b_frag(const _Float16* sB, int nbase, int lane) {
  HFrag f;
  int n  = lane & 15;
  int kg = (lane >> 4) * 16;
  const unsigned* rp = (const unsigned*)(sB + (nbase + n) * 32 + kg);
#pragma unroll
  for (int p = 0; p < 8; ++p) f.u[p] = rp[p];
  return f.v;
}

// ---------------------------------------------------------------------------
// Tensor Data Mover: DMA a <rows> x 32 f16 tile (row stride strideElems) from
// global into LDS at byte offset ldsOff, row-major. Rows >= validRows read as
// zero via the D# OOB rule (tensor_dim1). 2D tensor -> D# groups 0 and 1 only.
// (CDNA5 ISA) Tracked by TENSORcnt, in-order per wave.
// `ldsDst` is passed into the asm so the compiler knows LDS is written here
// (otherwise the never-stored LDS global folds the fragment loads to undef).
// ---------------------------------------------------------------------------
__device__ __forceinline__ void tdm_load_tile(const _Float16* gptr, void* ldsDst,
                                              unsigned ldsOff,
                                              int rows, int validRows, int strideElems) {
  unsigned long long ga = (unsigned long long)gptr;
  // D# group 0: [1:0]=count=1, [63:32]=lds_addr, [120:64]=global_addr,
  // [127:126]=type=2
  u32x4 g0 = { 1u,
               ldsOff,
               (unsigned)(ga & 0xffffffffu),
               (unsigned)((ga >> 32) & 0x01ffffffu) | 0x80000000u };
  // D# group 1: data_size=1 (2B) [17:16]; tensor_dim0=32 [79:48];
  // tensor_dim1=validRows [111:80]; tile_dim0=32 [127:112];
  // tile_dim1=rows [143:128]; tensor_dim0_stride=strideElems [207:160]
  u32x8 g1 = { 0x00010000u,
               32u << 16,                                   // tensor_dim0 lo16
               ((unsigned)validRows & 0xffffu) << 16,       // dim0 hi=0 | dim1 lo16
               (((unsigned)validRows >> 16) & 0xffffu) | (32u << 16), // dim1 hi | tile_dim0
               (unsigned)rows,                              // tile_dim1, tile_dim2=0
               (unsigned)strideElems,                       // dim0_stride lo32
               0u, 0u };
  // %2 (LDS dest pointer) is not printed but escapes into the asm: the TDM is
  // a potential writer of the shared tile, so LDS reads cannot be folded away.
  asm volatile("tensor_load_to_lds %0, %1"
               :: "s"(g0), "s"(g1), "s"(ldsDst) : "memory");
}

// ---------------------------------------------------------------------------
// Generic batched WMMA GEMM: C[M,N] (+resid) = A[M,K](f16,row-major) *
// Bw[N,K](f16, N-major / K-contiguous -- the native layout of all weights).
// Block tile 128x64, 8 waves, each wave 32x32. Both tiles staged by TDM with
// double buffering; compute waves issue no global loads in the hot loop.
// M multiple of 128; K multiple of 32; N handled via TDM OOB zero-fill +
// store guards.
// ---------------------------------------------------------------------------
__global__ __launch_bounds__(256) void wmma_gemm_kernel(
    const _Float16* __restrict__ A, const _Float16* __restrict__ Bw,
    float* __restrict__ C, const float* __restrict__ resid,
    int M, int N, int Kd,
    long long aStride, long long bStride, int bMod, long long cStride, int nTiles)
{
  // a[buf]: 128x32 tiles (8KB each @ 0/8192), b[buf]: 64x32 tiles (4KB each
  // @ 16384/20480), trailing guard word for the volatile store below.
  __shared__ struct { _Float16 a[2][128 * 32]; _Float16 b[2][64 * 32]; unsigned guard; } sm;

  int batch = blockIdx.y;
  A  += (long long)batch * aStride;
  Bw += (long long)(bMod ? (batch % bMod) : batch) * bStride;
  long long coff = (long long)batch * cStride;

  int tile = blockIdx.x;
  int tm = (tile / nTiles) * 128;
  int tn = (tile % nTiles) * 64;
  int tid  = threadIdx.x;
  int lane = tid & 31;
  int wave = tid >> 5;
  int wm = (wave >> 1) * 32;
  int wn = (wave & 1) * 32;

  // Visible, un-eliminable store to the LDS global (defeats "no stores ->
  // loads are undef" global optimization).
  if (tid == 0) *(volatile unsigned*)&sm.guard = (unsigned)N;

  const _Float16* Atile = A + (long long)tm * Kd;
  const _Float16* Btile = Bw + (long long)tn * Kd;
  int bValid = N - tn;                  // rows beyond this come back as zero

  v8f acc00 = {}, acc01 = {}, acc10 = {}, acc11 = {};

  int nsteps = Kd >> 5;
  if (wave == 0) {                      // prologue: fill buffer 0
    tdm_load_tile(Atile, (void*)&sm.a[0][0], 0u, 128, 128, Kd);
    tdm_load_tile(Btile, (void*)&sm.b[0][0], 16384u, 64, bValid, Kd);
  }

  for (int s = 0; s < nsteps; ++s) {
    int cur = s & 1;
    if (s + 1 < nsteps) {
      if (wave == 0) {                  // issue next K-slab into other buffer
        int k1 = (s + 1) << 5;
        unsigned nb = (unsigned)((s + 1) & 1);
        tdm_load_tile(Atile + k1, (void*)&sm.a[nb][0], nb * 8192u, 128, 128, Kd);
        tdm_load_tile(Btile + k1, (void*)&sm.b[nb][0], 16384u + nb * 4096u, 64, bValid, Kd);
        // 2 just issued may be in flight; the 2 for `cur` must have landed
        __builtin_amdgcn_s_wait_tensorcnt(2);
      }
    } else {
      if (wave == 0) __builtin_amdgcn_s_wait_tensorcnt(0);
    }
    __syncthreads();                    // publish buffer `cur`

    v16h a0 = load_a_frag(sm.a[cur], wm,      lane);
    v16h a1 = load_a_frag(sm.a[cur], wm + 16, lane);
    v16h b0 = load_b_frag(sm.b[cur], wn,      lane);
    v16h b1 = load_b_frag(sm.b[cur], wn + 16, lane);

    acc00 = __builtin_amdgcn_wmma_f32_16x16x32_f16(false, a0, false, b0, (short)0, acc00, false, false);
    acc01 = __builtin_amdgcn_wmma_f32_16x16x32_f16(false, a0, false, b1, (short)0, acc01, false, false);
    acc10 = __builtin_amdgcn_wmma_f32_16x16x32_f16(false, a1, false, b0, (short)0, acc10, false, false);
    acc11 = __builtin_amdgcn_wmma_f32_16x16x32_f16(false, a1, false, b1, (short)0, acc11, false, false);
    __syncthreads();   // reads of `cur` done before TDM overwrites it (s+2)
  }

  // C/D layout: VGPR r -> row r (lanes 0-15) / r+8 (lanes 16-31), col = lane&15
  float* Cb = C + coff;
  int cn = lane & 15;
  int rm = (lane >> 4) * 8;
#pragma unroll
  for (int r = 0; r < 8; ++r) {
    int r0 = tm + wm + rm + r;
    int r1 = r0 + 16;
    int c0 = tn + wn + cn;
    int c1 = c0 + 16;
    if (c0 < N) {
      float v0 = acc00[r]; if (resid) v0 += resid[coff + (long long)r0 * N + c0];
      Cb[(long long)r0 * N + c0] = v0;
      float v2 = acc10[r]; if (resid) v2 += resid[coff + (long long)r1 * N + c0];
      Cb[(long long)r1 * N + c0] = v2;
    }
    if (c1 < N) {
      float v1 = acc01[r]; if (resid) v1 += resid[coff + (long long)r0 * N + c1];
      Cb[(long long)r0 * N + c1] = v1;
      float v3 = acc11[r]; if (resid) v3 += resid[coff + (long long)r1 * N + c1];
      Cb[(long long)r1 * N + c1] = v3;
    }
  }
}

// ---------------------------------------------------------------------------
// AdaRMSNorm scale: scale[b,c] = dot(cond[b,:], w_ada[c,:]) + 1
// ---------------------------------------------------------------------------
__global__ void ada_scale_kernel(const float* __restrict__ cond,
                                 const float* __restrict__ w_ada,
                                 float* __restrict__ scale) {
  int b = blockIdx.x, c = threadIdx.x;
  float s = 0.f;
  for (int j = 0; j < DM; ++j) s += cond[b * DM + j] * w_ada[c * DM + j];
  scale[b * DM + c] = s + 1.f;
}

// RMSNorm over channels, emit f16 activations for WMMA
__global__ __launch_bounds__(256) void rmsnorm_kernel(const float* __restrict__ x,
                                                      const float* __restrict__ scale,
                                                      _Float16* __restrict__ xn) {
  __shared__ float s1[256];
  long long row = blockIdx.x;
  int c = threadIdx.x;
  int b = (int)(row >> 12);
  float v = x[row * DM + c];
  s1[c] = v * v;
  __syncthreads();
  for (int off = 128; off > 0; off >>= 1) {
    if (c < off) s1[c] += s1[c + off];
    __syncthreads();
  }
  float r = rsqrtf(s1[0] * (1.f / DM) + 1e-6f);
  xn[row * DM + c] = (_Float16)(v * scale[b * DM + c] * r);
}

// ---------------------------------------------------------------------------
// Weight conversion: all weights are natively [N, K] (K contiguous) which is
// exactly the TDM B-tile layout -- pure f32 -> f16 copy, no transpose.
// ---------------------------------------------------------------------------
__global__ void cvt_f16_kernel(const float* __restrict__ w, _Float16* __restrict__ o, int n) {
  int i = blockIdx.x * 256 + threadIdx.x;
  if (i < n) o[i] = (_Float16)w[i];
}
// dt_proj_w [k][d][r=16] -> [k][d][r=32] zero-padded so GEMM-K is 32
__global__ void pack_dtw_kernel(const float* __restrict__ w, _Float16* __restrict__ o) {
  int i = blockIdx.x * 256 + threadIdx.x;     // i = (k*512 + d)*32 + r
  int r = i & 31;
  int kd = i >> 5;
  o[i] = (_Float16)(r < DTR ? w[kd * DTR + r] : 0.f);
}

// ---------------------------------------------------------------------------
// Depthwise 3x3 conv + SiLU, scatter into the 4 cross-scan directions (f16)
// ---------------------------------------------------------------------------
__global__ __launch_bounds__(512) void conv_scan_kernel(const float* __restrict__ xz,
                                                        const float* __restrict__ cw,
                                                        const float* __restrict__ cb,
                                                        _Float16* __restrict__ xs) {
  int d = threadIdx.x;
  int sp = blockIdx.x;
  int b = sp >> 12;
  int hw = sp & 4095;
  int hh = hw >> 6, ww = hw & 63;
  float acc = cb[d];
#pragma unroll
  for (int dy = 0; dy < 3; ++dy) {
    int yy = hh + dy - 1;
    if ((unsigned)yy >= 64u) continue;
#pragma unroll
    for (int dx = 0; dx < 3; ++dx) {
      int xw = ww + dx - 1;
      if ((unsigned)xw >= 64u) continue;
      acc += xz[((long long)(b * LL + yy * WW_ + xw)) * (2 * DI) + d] * cw[d * 9 + dy * 3 + dx];
    }
  }
  float v = acc / (1.f + __expf(-acc));   // SiLU
  _Float16 hv = (_Float16)v;
  long long bb = (long long)b * KK * LL * DI;
  int l0 = hw;                 // hw order
  int l1 = ww * 64 + hh;       // wh order
  xs[bb + ((long long)(0 * LL + l0)) * DI + d] = hv;
  xs[bb + ((long long)(1 * LL + l1)) * DI + d] = hv;
  xs[bb + ((long long)(2 * LL + (LL - 1 - l0))) * DI + d] = hv;
  xs[bb + ((long long)(3 * LL + (LL - 1 - l1))) * DI + d] = hv;
}

// dts: cols 0..15 of x_dbl, zero-padded to 32 for K-alignment of the dt GEMM
__global__ void pack_dts_kernel(const float* __restrict__ xdbl, _Float16* __restrict__ dts) {
  long long i = (long long)blockIdx.x * 256 + threadIdx.x;  // BK*L*32
  int r = (int)(i & 31);
  long long bl = i >> 5;
  dts[i] = (_Float16)(r < DTR ? xdbl[bl * CPROJ + r] : 0.f);
}

// ---------------------------------------------------------------------------
// Selective scan. One block per (b,k), one channel per thread, 16 states in
// registers, B/C broadcast via double-buffered LDS (one barrier per step).
// ys is written IN PLACE over the consumed dt buffer at scan position l.
// ---------------------------------------------------------------------------
__global__ __launch_bounds__(512) void scan_kernel(const float* __restrict__ xdbl,
                                                   float* __restrict__ dtys,
                                                   const _Float16* __restrict__ xs,
                                                   const float* __restrict__ A_logs,
                                                   const float* __restrict__ Ds,
                                                   const float* __restrict__ dt_b) {
  int bk = blockIdx.x;
  int k = bk & 3;
  int d = threadIdx.x;
  int kd = k * DI + d;
  float A[NS];
#pragma unroll
  for (int n = 0; n < NS; ++n) A[n] = -__expf(A_logs[kd * NS + n]);
  float Dv = Ds[kd];
  float bias = dt_b[kd];
  float h[NS];
#pragma unroll
  for (int n = 0; n < NS; ++n) h[n] = 0.f;

  __shared__ float sBC[2][2 * NS];
  const long long base48  = (long long)bk * LL * CPROJ;
  const long long base512 = (long long)bk * LL * DI;

  for (int l = 0; l < LL; ++l) {
    int p = l & 1;
    if (d < 2 * NS) sBC[p][d] = xdbl[base48 + (long long)l * CPROJ + DTR + d];
    __syncthreads();
    float dtr = dtys[base512 + (long long)l * DI + d] + bias;
    float dt = (dtr > 20.f) ? dtr : log1pf(__expf(dtr));   // softplus
    float u = (float)xs[base512 + (long long)l * DI + d];
    float du = dt * u;
    float y = Dv * u;
#pragma unroll
    for (int n = 0; n < NS; ++n) {
      h[n] = h[n] * __expf(dt * A[n]) + du * sBC[p][n];
      y += h[n] * sBC[p][NS + n];
    }
    dtys[base512 + (long long)l * DI + d] = y;   // in-place ys
  }
}

// ---------------------------------------------------------------------------
// Deterministic cross-merge (4-way gather) + LayerNorm + SiLU gate -> f16
// ---------------------------------------------------------------------------
__global__ __launch_bounds__(512) void ln_gate_kernel(const float* __restrict__ ys,
                                                      const float* __restrict__ xz,
                                                      const float* __restrict__ ln_w,
                                                      const float* __restrict__ ln_b,
                                                      _Float16* __restrict__ yg) {
  __shared__ float s1[512];
  __shared__ float s2[512];
  int row = blockIdx.x;            // b*L + pos (hw order)
  int b = row >> 12;
  int pos = row & 4095;
  int hh = pos >> 6, ww = pos & 63;
  int l1 = ww * 64 + hh;           // wh order index
  int d = threadIdx.x;
  long long base = (long long)b * KK * LL * DI;
  float v = ys[base + (long long)(0 * LL + pos) * DI + d]
          + ys[base + (long long)(1 * LL + l1) * DI + d]
          + ys[base + (long long)(2 * LL + (LL - 1 - pos)) * DI + d]
          + ys[base + (long long)(3 * LL + (LL - 1 - l1)) * DI + d];
  s1[d] = v; s2[d] = v * v;
  __syncthreads();
  for (int off = 256; off > 0; off >>= 1) {
    if (d < off) { s1[d] += s1[d + off]; s2[d] += s2[d + off]; }
    __syncthreads();
  }
  float mu = s1[0] * (1.f / DI);
  float var = s2[0] * (1.f / DI) - mu * mu;
  float yn = (v - mu) * rsqrtf(var + 1e-5f) * ln_w[d] + ln_b[d];
  float z = xz[(long long)row * (2 * DI) + DI + d];
  float sz = z / (1.f + __expf(-z));
  yg[(long long)row * DI + d] = (_Float16)(yn * sz);
}

// ---------------------------------------------------------------------------
// Host launcher
// ---------------------------------------------------------------------------
extern "C" void kernel_launch(void* const* d_in, const int* in_sizes, int n_in,
                              void* d_out, int out_size, void* d_ws, size_t ws_size,
                              hipStream_t stream) {
  (void)in_sizes; (void)n_in; (void)out_size; (void)ws_size;

  const float* x         = (const float*)d_in[0];
  /* d_in[1] = pos, unused */
  const float* cond      = (const float*)d_in[2];
  const float* w_ada     = (const float*)d_in[3];
  const float* w_in      = (const float*)d_in[4];
  const float* conv_w    = (const float*)d_in[5];
  const float* conv_b    = (const float*)d_in[6];
  const float* x_proj_w  = (const float*)d_in[7];
  const float* dt_proj_w = (const float*)d_in[8];
  const float* dt_proj_b = (const float*)d_in[9];
  const float* A_logs    = (const float*)d_in[10];
  const float* Ds        = (const float*)d_in[11];
  const float* ln_w      = (const float*)d_in[12];
  const float* ln_b      = (const float*)d_in[13];
  const float* w_out     = (const float*)d_in[14];

  char* ws = (char*)d_ws;
  size_t o = 0;
  auto take = [&](size_t bytes) -> char* {
    char* p = ws + o;
    o += (bytes + 255) & ~(size_t)255;
    return p;
  };

  float*     scale = (float*)    take((size_t)BB * DM * 4);
  _Float16*  xn    = (_Float16*) take((size_t)MROWS * DM * 2);
  _Float16*  w_inB = (_Float16*) take((size_t)2 * DI * DM * 2);          // [1024,256]
  float*     xz    = (float*)    take((size_t)MROWS * 2 * DI * 4);
  _Float16*  xs    = (_Float16*) take((size_t)BB * KK * LL * DI * 2);
  _Float16*  xpwB  = (_Float16*) take((size_t)KK * CPROJ * DI * 2);      // [k][48,512]
  float*     xdbl  = (float*)    take((size_t)BB * KK * LL * CPROJ * 4);
  _Float16*  dts   = (_Float16*) take((size_t)BB * KK * LL * 32 * 2);
  _Float16*  dtwB  = (_Float16*) take((size_t)KK * DI * 32 * 2);         // [k][512,32]
  float*     dtys  = (float*)    take((size_t)BB * KK * LL * DI * 4);    // dt, then ys in place
  _Float16*  yg    = (_Float16*) take((size_t)MROWS * DI * 2);
  _Float16*  woutB = (_Float16*) take((size_t)DM * DI * 2);              // [256,512]

  // 1) AdaRMSNorm scale + weight converts (all weights stay native [N,K])
  ada_scale_kernel<<<BB, DM, 0, stream>>>(cond, w_ada, scale);
  cvt_f16_kernel<<<(2 * DI * DM) / 256, 256, 0, stream>>>(w_in, w_inB, 2 * DI * DM);
  cvt_f16_kernel<<<(KK * CPROJ * DI) / 256, 256, 0, stream>>>(x_proj_w, xpwB, KK * CPROJ * DI);
  pack_dtw_kernel<<<(KK * DI * 32) / 256, 256, 0, stream>>>(dt_proj_w, dtwB);
  cvt_f16_kernel<<<(DM * DI) / 256, 256, 0, stream>>>(w_out, woutB, DM * DI);

  // 2) RMSNorm -> f16 activations
  rmsnorm_kernel<<<MROWS, DM, 0, stream>>>(x, scale, xn);

  // 3) in_proj GEMM: [16384,1024] = xn[16384,256] x w_inB[1024,256]^T
  wmma_gemm_kernel<<<dim3((MROWS / 128) * (2 * DI / 64), 1), 256, 0, stream>>>(
      xn, w_inB, xz, nullptr, MROWS, 2 * DI, DM, 0, 0, 0, 0, 2 * DI / 64);

  // 4) depthwise conv + SiLU + cross-scan scatter
  conv_scan_kernel<<<MROWS, DI, 0, stream>>>(xz, conv_w, conv_b, xs);

  // 5) x_proj GEMM per (b,k): [4096,48] = xs[4096,512] x xpwB_k[48,512]^T
  wmma_gemm_kernel<<<dim3(LL / 128, BB * KK), 256, 0, stream>>>(
      xs, xpwB, xdbl, nullptr, LL, CPROJ, DI,
      (long long)LL * DI, (long long)CPROJ * DI, KK, (long long)LL * CPROJ, 1);

  // 6) extract+pad dts to K=32
  pack_dts_kernel<<<(BB * KK * LL * 32) / 256, 256, 0, stream>>>(xdbl, dts);

  // 7) dt_proj GEMM per (b,k): [4096,512] = dts[4096,32] x dtwB_k[512,32]^T
  wmma_gemm_kernel<<<dim3((LL / 128) * (DI / 64), BB * KK), 256, 0, stream>>>(
      dts, dtwB, dtys, nullptr, LL, DI, 32,
      (long long)LL * 32, (long long)DI * 32, KK, (long long)LL * DI, DI / 64);

  // 8) selective scan (ys overwrites dtys in place)
  scan_kernel<<<BB * KK, DI, 0, stream>>>(xdbl, dtys, xs, A_logs, Ds, dt_proj_b);

  // 9) deterministic cross-merge + LayerNorm + gate -> f16
  ln_gate_kernel<<<MROWS, DI, 0, stream>>>(dtys, xz, ln_w, ln_b, yg);

  // 10) out_proj GEMM + residual: out[16384,256] = yg[16384,512] x woutB[256,512]^T + x
  wmma_gemm_kernel<<<dim3((MROWS / 128) * (DM / 64), 1), 256, 0, stream>>>(
      yg, woutB, (float*)d_out, x, MROWS, DM, DI, 0, 0, 0, 0, DM / 64);
}
